// FastSpeech2_10840497455834
// MI455X (gfx1250) — compile-verified, matrix operand order checked by reading, not run on hardware
//
#include <hip/hip_runtime.h>
#include <hip/hip_bf16.h>
#include <math.h>

// ---- model constants (match reference) ----
#define BB      4
#define SS      512
#define MAXL    2048
#define HD      256
#define NHD     2
#define DKK     (HD / NHD)        // 128
#define FFTF    1024
#define NMELS   80

typedef __attribute__((ext_vector_type(16))) _Float16 v16h;
typedef __attribute__((ext_vector_type(8)))  _Float16 v8h;
typedef __attribute__((ext_vector_type(8)))  float    v8f;

// Per CDNA5 ISA 7.12.2: a 16x32 f16 A/B fragment's 16 halves per lane are two
// contiguous K runs: [k0 + hi*8, +8) and [k0 + 16 + hi*8, +8), hi = lane>>4.
#define SHUF16(a, b) __builtin_shufflevector(a, b, 0,1,2,3,4,5,6,7,8,9,10,11,12,13,14,15)

// ---------------------------------------------------------------------------
// Weight pack: Wp[dt][n][k] (f16, k-contiguous) from W[n*wsN + k*wsK + dt*wsD]
// ---------------------------------------------------------------------------
__global__ void pack_w_k(const float* __restrict__ W, _Float16* __restrict__ Wp,
                         int N, int K, long wsN, long wsK, long wsD)
{
  const long i  = (long)blockIdx.x * blockDim.x + threadIdx.x;   // over KW*N*K
  const long k  = i % K;
  const long n  = (i / K) % N;
  const long dt = i / ((long)N * K);
  Wp[i] = (_Float16)W[n * wsN + k * wsK + dt * wsD];
}

// ---------------------------------------------------------------------------
// WMMA GEMM with fused shifted-conv / bias / relu. All-f16 operand streams.
//   C[m,n] = sum_{dt,k} A[b, t+dt-pad, k] * Wp[dt][n][k]   (+bias, relu)
// A: (M,K) f16 row-major, M = BB*T; rows outside [0,T) per batch contribute 0.
// One wave computes a 16 x (16*NT) tile: NT independent accumulators so the
// NT v_wmma per K-step pipeline back-to-back; the A fragment is reused NT x.
// Interior tiles (the vast majority) take a branch-free, mask-free fast path.
// Outputs: Cf (f32) and/or Ch (f16 mirror) — either may be null.
// ---------------------------------------------------------------------------
template <int NT>
__global__ __launch_bounds__(32) void wmma_gemm_t(
    const _Float16* __restrict__ A, const _Float16* __restrict__ Wp,
    const float* __restrict__ bias, float* __restrict__ Cf,
    _Float16* __restrict__ Ch,
    int M, int N, int K, int T, int KW, int pad, int relu)
{
  const int lane  = threadIdx.x;
  const int laneN = lane & 15;
  const int hi    = lane >> 4;
  const int mt = blockIdx.x, nt = blockIdx.y;

  const int t0    = (mt * 16) % T;           // tile row start within batch (uniform)
  const int b     = (mt * 16) / T;           // batch (uniform; tiles never straddle)
  const int t     = t0 + laneN;              // A-fragment row for this lane
  const int ncol0 = nt * (16 * NT) + laneN;  // first B-fragment column for this lane

  // all rows t0..t0+15 valid for every dt in [0,KW)?
  const bool interior = (t0 - pad >= 0) && (t0 + 15 + (KW - 1) - pad <= T - 1);

  v8f acc[NT];
#pragma unroll
  for (int j = 0; j < NT; ++j) acc[j] = (v8f){};

  if (interior) {
    const _Float16* arow0 = A + (long)(b * T + t - pad) * K + hi * 8;
    for (int dt = 0; dt < KW; ++dt) {
      const _Float16* arow = arow0 + (long)dt * K;
      const _Float16* wrow = Wp + ((long)dt * N + ncol0) * (long)K + hi * 8;
      for (int k0 = 0; k0 < K; k0 += 32) {
        __builtin_prefetch(arow + k0 + 64, 0, 0);       // -> global_prefetch
        const v8h a0 = *(const v8h*)(arow + k0);
        const v8h a1 = *(const v8h*)(arow + k0 + 16);
        const v16h af = SHUF16(a0, a1);
#pragma unroll
        for (int j = 0; j < NT; ++j) {
          const _Float16* wj = wrow + (long)j * 16 * K + k0;
          const v8h b0 = *(const v8h*)(wj);
          const v8h b1 = *(const v8h*)(wj + 16);
          const v16h bf = SHUF16(b0, b1);
          acc[j] = __builtin_amdgcn_wmma_f32_16x16x32_f16(
              false, af, false, bf, (short)0, acc[j], false, false);
        }
      }
    }
  } else {
    for (int dt = 0; dt < KW; ++dt) {
      const int tt = t + dt - pad;
      const _Float16 hmsk = (_Float16)((tt >= 0 && tt < T) ? 1.0f : 0.0f);
      const int ttc = (tt < 0) ? 0 : ((tt >= T) ? (T - 1) : tt);
      const _Float16* arow = A + (long)(b * T + ttc) * K + hi * 8;
      const _Float16* wrow = Wp + ((long)dt * N + ncol0) * (long)K + hi * 8;
      for (int k0 = 0; k0 < K; k0 += 32) {
        const v8h a0 = *(const v8h*)(arow + k0);
        const v8h a1 = *(const v8h*)(arow + k0 + 16);
        v16h af;
#pragma unroll
        for (int e = 0; e < 8; ++e) { af[e] = a0[e] * hmsk; af[e + 8] = a1[e] * hmsk; }
#pragma unroll
        for (int j = 0; j < NT; ++j) {
          const _Float16* wj = wrow + (long)j * 16 * K + k0;
          const v8h b0 = *(const v8h*)(wj);
          const v8h b1 = *(const v8h*)(wj + 16);
          const v16h bf = SHUF16(b0, b1);
          acc[j] = __builtin_amdgcn_wmma_f32_16x16x32_f16(
              false, af, false, bf, (short)0, acc[j], false, false);
        }
      }
    }
  }

#pragma unroll
  for (int j = 0; j < NT; ++j) {
    const int ncol = ncol0 + j * 16;
    const float bv = bias ? bias[ncol] : 0.0f;
#pragma unroll
    for (int r = 0; r < 8; ++r) {
      const int m = mt * 16 + r + hi * 8;
      float v = acc[j][r] + bv;
      if (relu) v = fmaxf(v, 0.0f);
      if (Cf) Cf[(long)m * N + ncol] = v;
      if (Ch) Ch[(long)m * N + ncol] = (_Float16)v;
    }
  }
}

// ---------------------------------------------------------------------------
// Flash attention: one wave per (b, head, 16-query tile). dk = 128.
// Q,K,V are rows of a fused (BB, T, ldx) f16 buffer (ldx = 768 for fused QKV);
// O is (BB, T, HD) f16; mask is (BB, T) bool over keys.
// ---------------------------------------------------------------------------
__global__ __launch_bounds__(32) void flash_attn_k(
    const _Float16* __restrict__ Qb, const _Float16* __restrict__ Kb,
    const _Float16* __restrict__ Vb, _Float16* __restrict__ Ob,
    const unsigned char* __restrict__ mask, int T, int ldx, float scale)
{
  const int lane  = threadIdx.x;
  const int laneN = lane & 15;
  const int hi    = lane >> 4;
  const int qt   = blockIdx.x;
  const int head = blockIdx.y;
  const int b    = blockIdx.z;
  const int hoff = head * DKK;
  const long rowbase = (long)b * T;

  // Preload Q tile (16 x 128) as four 16x32 A fragments (pure v8h loads).
  v16h qa[4];
  {
    const _Float16* qrow = Qb + (rowbase + qt * 16 + laneN) * ldx + hoff + hi * 8;
#pragma unroll
    for (int c = 0; c < 4; ++c) {
      const v8h q0 = *(const v8h*)(qrow + c * 32);
      const v8h q1 = *(const v8h*)(qrow + c * 32 + 16);
      qa[c] = SHUF16(q0, q1);
    }
  }

  v8f o[8];
#pragma unroll
  for (int i = 0; i < 8; ++i) o[i] = (v8f){};
  float rmax[8], rsum[8];
#pragma unroll
  for (int r = 0; r < 8; ++r) { rmax[r] = -3.0e38f; rsum[r] = 0.0f; }

  __shared__ _Float16 plds[16 * 32];   // one wave per block

  for (int kc = 0; kc < T; kc += 32) {
    // scores for 32 keys as two 16x16 C tiles
    v8f s0 = {}, s1 = {};
    const _Float16* k0r = Kb + (rowbase + kc + laneN) * ldx + hoff + hi * 8;
    const _Float16* k1r = Kb + (rowbase + kc + 16 + laneN) * ldx + hoff + hi * 8;
#pragma unroll
    for (int c = 0; c < 4; ++c) {
      const v8h x0 = *(const v8h*)(k0r + c * 32);
      const v8h x1 = *(const v8h*)(k0r + c * 32 + 16);
      const v8h y0 = *(const v8h*)(k1r + c * 32);
      const v8h y1 = *(const v8h*)(k1r + c * 32 + 16);
      const v16h b0 = SHUF16(x0, x1);
      const v16h b1 = SHUF16(y0, y1);
      s0 = __builtin_amdgcn_wmma_f32_16x16x32_f16(false, qa[c], false, b0, (short)0, s0, false, false);
      s1 = __builtin_amdgcn_wmma_f32_16x16x32_f16(false, qa[c], false, b1, (short)0, s1, false, false);
    }
    const bool mk0 = mask[rowbase + kc + laneN] != 0;
    const bool mk1 = mask[rowbase + kc + 16 + laneN] != 0;

#pragma unroll
    for (int r = 0; r < 8; ++r) {
      const float v0 = mk0 ? -1e9f : s0[r] * scale;
      const float v1 = mk1 ? -1e9f : s1[r] * scale;
      float tmx = fmaxf(v0, v1);
      tmx = fmaxf(tmx, __shfl_xor(tmx, 1, 32));
      tmx = fmaxf(tmx, __shfl_xor(tmx, 2, 32));
      tmx = fmaxf(tmx, __shfl_xor(tmx, 4, 32));
      tmx = fmaxf(tmx, __shfl_xor(tmx, 8, 32));
      const float nm = fmaxf(rmax[r], tmx);
      const float al = __expf(rmax[r] - nm);
      const float p0 = __expf(v0 - nm);
      const float p1 = __expf(v1 - nm);
      float rs = p0 + p1;
      rs += __shfl_xor(rs, 1, 32);
      rs += __shfl_xor(rs, 2, 32);
      rs += __shfl_xor(rs, 4, 32);
      rs += __shfl_xor(rs, 8, 32);
      rsum[r] = rsum[r] * al + rs;
      rmax[r] = nm;
#pragma unroll
      for (int n2 = 0; n2 < 8; ++n2) o[n2][r] *= al;
      const int row = r + 8 * hi;
      plds[row * 32 + laneN]      = (_Float16)p0;
      plds[row * 32 + 16 + laneN] = (_Float16)p1;
    }
    __syncthreads();

    // reload P (16x32) as an A fragment and accumulate O += P @ V
    v16h pa;
#pragma unroll
    for (int e = 0; e < 8; ++e) {
      pa[e]     = plds[laneN * 32 + hi * 8 + e];
      pa[e + 8] = plds[laneN * 32 + 16 + hi * 8 + e];
    }
#pragma unroll
    for (int n2 = 0; n2 < 8; ++n2) {
      v16h vb;
      const _Float16* vcol = Vb + (rowbase + kc) * ldx + hoff + n2 * 16 + laneN;
#pragma unroll
      for (int e = 0; e < 8; ++e) {
        vb[e]     = vcol[(hi * 8 + e) * ldx];
        vb[e + 8] = vcol[(16 + hi * 8 + e) * ldx];
      }
      o[n2] = __builtin_amdgcn_wmma_f32_16x16x32_f16(false, pa, false, vb, (short)0, o[n2], false, false);
    }
    __syncthreads();
  }

#pragma unroll
  for (int n2 = 0; n2 < 8; ++n2)
#pragma unroll
    for (int r = 0; r < 8; ++r) {
      const int m = qt * 16 + r + 8 * hi;
      Ob[(rowbase + m) * HD + hoff + n2 * 16 + laneN] = (_Float16)(o[n2][r] / rsum[r]);
    }
}

// ---- LayerNorm over last dim (H=256); fused residual; writes f32 + f16 mirror ----
__global__ void layernorm_k(const float* __restrict__ x, const float* __restrict__ res,
                            const float* __restrict__ g, const float* __restrict__ bt,
                            float* __restrict__ out, _Float16* __restrict__ out16, int H)
{
  const int m = blockIdx.x, c = threadIdx.x;
  __shared__ float s1[256], s2[256];
  float v = x[(long)m * H + c];
  if (res) v += res[(long)m * H + c];
  s1[c] = v; s2[c] = v * v;
  __syncthreads();
  for (int st = 128; st > 0; st >>= 1) {
    if (c < st) { s1[c] += s1[c + st]; s2[c] += s2[c + st]; }
    __syncthreads();
  }
  const float mean = s1[0] / H;
  const float var  = s2[0] / H - mean * mean;
  const float y = (v - mean) * rsqrtf(var + 1e-5f) * g[c] + bt[c];
  out[(long)m * H + c]   = y;
  out16[(long)m * H + c] = (_Float16)y;
}

// ---- GroupNorm(num_groups=1): per-batch stats over (T,C), then apply ----
__global__ void gn_stats_k(const float* __restrict__ x, float* __restrict__ stats, long cnt)
{
  const int b = blockIdx.x, tid = threadIdx.x;
  const float* p = x + (long)b * cnt;
  float s = 0.f, sq = 0.f;
  for (long i = tid; i < cnt; i += 256) { const float v = p[i]; s += v; sq += v * v; }
  __shared__ float a1[256], a2[256];
  a1[tid] = s; a2[tid] = sq;
  __syncthreads();
  for (int st = 128; st > 0; st >>= 1) {
    if (tid < st) { a1[tid] += a1[tid + st]; a2[tid] += a2[tid + st]; }
    __syncthreads();
  }
  if (tid == 0) {
    const float mean = a1[0] / (float)cnt;
    stats[b * 2]     = mean;
    stats[b * 2 + 1] = a2[0] / (float)cnt - mean * mean;
  }
}

__global__ void gn_apply_k(float* __restrict__ x, const float* __restrict__ stats,
                           const float* __restrict__ g, const float* __restrict__ bt,
                           _Float16* __restrict__ x16, int C, long perB)
{
  const long i = (long)blockIdx.x * blockDim.x + threadIdx.x;
  const int b = (int)(i / perB);
  const int c = (int)(i % C);
  const float m = stats[b * 2], v = stats[b * 2 + 1];
  const float y = (x[i] - m) * rsqrtf(v + 1e-5f) * g[c] + bt[c];
  x[i] = y;
  if (x16) x16[i] = (_Float16)y;
}

// ---- variance-predictor head: out[m] = mask? 0 : h[m,:] . w + b ----
__global__ void vp_linear_k(const float* __restrict__ h, const float* __restrict__ w,
                            const float* __restrict__ bias,
                            const unsigned char* __restrict__ mask,
                            float* __restrict__ out, int K)
{
  const int m = blockIdx.x, c = threadIdx.x;
  __shared__ float s[256];
  s[c] = h[(long)m * K + c] * w[c];
  __syncthreads();
  for (int st = 128; st > 0; st >>= 1) {
    if (c < st) s[c] += s[c + st];
    __syncthreads();
  }
  if (c == 0) out[m] = mask[m] ? 0.0f : (s[0] + bias[0]);
}

// ---- embedding gather (writes f32 + f16 mirror) ----
__global__ void embed_k(const int* __restrict__ seq, const float* __restrict__ emb,
                        float* __restrict__ out, _Float16* __restrict__ out16, int H)
{
  const int m = blockIdx.x, c = threadIdx.x;
  const float v = emb[(long)seq[m] * H + c];
  out[(long)m * H + c]   = v;
  out16[(long)m * H + c] = (_Float16)v;
}

// ---- duration cumsum (tiny) ----
__global__ void cumsum_k(const int* __restrict__ dur, int* __restrict__ cum, int S)
{
  const int b = threadIdx.x;
  int acc = 0;
  for (int i = 0; i < S; ++i) { acc += dur[b * S + i]; cum[b * S + i] = acc; }
}

// ---- length regulate: frame t copies phoneme j = searchsorted(cum, t, 'right') ----
__global__ void length_reg_k(const float* __restrict__ x, const int* __restrict__ cum,
                             float* __restrict__ out, _Float16* __restrict__ out16,
                             int S, int H, int maxlen)
{
  const int t = blockIdx.x % maxlen;
  const int b = blockIdx.x / maxlen;
  const int c = threadIdx.x;
  const int* cb = cum + b * S;
  int lo = 0, hiB = S;
  while (lo < hiB) { const int mid = (lo + hiB) >> 1; if (cb[mid] <= t) lo = mid + 1; else hiB = mid; }
  const float v = (lo < S) ? x[((long)b * S + lo) * H + c] : 0.0f;
  out[((long)b * maxlen + t) * H + c]   = v;
  out16[((long)b * maxlen + t) * H + c] = (_Float16)v;
}

// ===========================================================================
// Host-side orchestration
// ===========================================================================
struct FFTP { const float *c1_b,*c1_w,*c2_b,*c2_w,*fc,*ln1_b,*ln1_s,*ln2_b,*ln2_s,*wk,*wq,*wv; };
struct VPP  { const float *c1_b,*c1_w,*c2_b,*c2_w,*gn1_b,*gn1_s,*gn2_b,*gn2_s,*lin_b,*lin_w; };

static void run_gemm(hipStream_t st, const _Float16* A, const float* W, const float* bias,
                     float* Cf, _Float16* Ch, _Float16* Wp,
                     int M, int N, int K, int T, int KW, int pad,
                     long wsN, long wsK, long wsD, int relu)
{
  const long tot = (long)KW * N * K;
  pack_w_k<<<(unsigned)(tot / 256), 256, 0, st>>>(W, Wp, N, K, wsN, wsK, wsD);
  if (N % 64 == 0) {
    dim3 g((unsigned)(M / 16), (unsigned)(N / 64));
    wmma_gemm_t<4><<<g, 32, 0, st>>>(A, Wp, bias, Cf, Ch, M, N, K, T, KW, pad, relu);
  } else {
    dim3 g((unsigned)(M / 16), (unsigned)(N / 16));
    wmma_gemm_t<1><<<g, 32, 0, st>>>(A, Wp, bias, Cf, Ch, M, N, K, T, KW, pad, relu);
  }
}

static void run_fft_block(hipStream_t st, float* X, _Float16* X16, int T,
                          const unsigned char* mask, const FFTP& p,
                          _Float16* QKV16, _Float16* O16,
                          float* Fb, _Float16* H16, _Float16* Wp)
{
  const int M = BB * T;
  // fused QKV projection: pack wq|wk|wv as one [n<768][k<256] image, one GEMM
  pack_w_k<<<(HD * HD) / 256, 256, 0, st>>>(p.wq, Wp,               HD, HD, 1L, (long)HD, 0L);
  pack_w_k<<<(HD * HD) / 256, 256, 0, st>>>(p.wk, Wp + 1 * HD * HD, HD, HD, 1L, (long)HD, 0L);
  pack_w_k<<<(HD * HD) / 256, 256, 0, st>>>(p.wv, Wp + 2 * HD * HD, HD, HD, 1L, (long)HD, 0L);
  {
    dim3 g((unsigned)(M / 16), (unsigned)(3 * HD / 64));
    wmma_gemm_t<4><<<g, 32, 0, st>>>(X16, Wp, nullptr, nullptr, QKV16,
                                     M, 3 * HD, HD, T, 1, 0, 0);
  }
  dim3 ga((unsigned)(T / 16), NHD, BB);
  flash_attn_k<<<ga, 32, 0, st>>>(QKV16, QKV16 + HD, QKV16 + 2 * HD, O16,
                                  mask, T, 3 * HD, 1.0f / sqrtf((float)DKK));
  run_gemm(st, O16, p.fc, nullptr, Fb, nullptr, Wp, M, HD, HD, T, 1, 0, 1L, (long)HD, 0L, 0);
  layernorm_k<<<M, HD, 0, st>>>(X, Fb, p.ln1_s, p.ln1_b, X, X16, HD);
  // conv1: (H -> FFTF), kernel 9, pad 4, relu; weights (FFTF, H, 9); f16-only hidden
  run_gemm(st, X16, p.c1_w, p.c1_b, nullptr, H16, Wp, M, FFTF, HD, T, 9, 4, (long)HD * 9, 9L, 1L, 1);
  // conv2: 1x1 (FFTF -> H); weights (H, FFTF, 1)
  run_gemm(st, H16, p.c2_w, p.c2_b, Fb, nullptr, Wp, M, HD, FFTF, T, 1, 0, (long)FFTF, 1L, 0L, 0);
  layernorm_k<<<M, HD, 0, st>>>(X, Fb, p.ln2_s, p.ln2_b, X, X16, HD);
}

static void run_vp(hipStream_t st, const _Float16* X16, const unsigned char* mask, const VPP& p,
                   float* Fb, float* Gb, _Float16* Th, float* gstats, float* out, _Float16* Wp)
{
  const int  M    = BB * SS;
  const long perB = (long)SS * HD;
  run_gemm(st, X16, p.c1_w, p.c1_b, Fb, nullptr, Wp, M, HD, HD, SS, 3, 1, (long)HD * 3, 3L, 1L, 1);
  gn_stats_k<<<BB, 256, 0, st>>>(Fb, gstats, perB);
  gn_apply_k<<<(unsigned)((BB * perB) / 256), 256, 0, st>>>(Fb, gstats, p.gn1_s, p.gn1_b, Th, HD, perB);
  run_gemm(st, Th, p.c2_w, p.c2_b, Gb, nullptr, Wp, M, HD, HD, SS, 3, 1, (long)HD * 3, 3L, 1L, 1);
  gn_stats_k<<<BB, 256, 0, st>>>(Gb, gstats, perB);
  gn_apply_k<<<(unsigned)((BB * perB) / 256), 256, 0, st>>>(Gb, gstats, p.gn2_s, p.gn2_b, nullptr, HD, perB);
  vp_linear_k<<<M, 256, 0, st>>>(Gb, p.lin_w, p.lin_b, mask, out, HD);
}

extern "C" void kernel_launch(void* const* d_in, const int* in_sizes, int n_in,
                              void* d_out, int out_size, void* d_ws, size_t ws_size,
                              hipStream_t stream)
{
  (void)in_sizes; (void)n_in; (void)out_size; (void)ws_size;

  const int*           src_seq  = (const int*)d_in[0];
  const unsigned char* src_mask = (const unsigned char*)d_in[1];
  const unsigned char* mel_mask = (const unsigned char*)d_in[2];
  const int*           dur_tgt  = (const int*)d_in[3];
  // d_in[4] = max_len scalar (== MAXL, fixed)

  // params flattened in JAX pytree order (dict keys sorted alphabetically):
  // top: dec(4x12), emb, enc(4x12), mel_b, mel_w, vp(3x10)
  const int PB = 5;
  auto FP = [&](int i) { return (const float*)d_in[i]; };
  auto fillF = [&](FFTP& q, int base) {
    q.c1_b = FP(base + 0);  q.c1_w = FP(base + 1);  q.c2_b = FP(base + 2);
    q.c2_w = FP(base + 3);  q.fc   = FP(base + 4);  q.ln1_b = FP(base + 5);
    q.ln1_s = FP(base + 6); q.ln2_b = FP(base + 7); q.ln2_s = FP(base + 8);
    q.wk = FP(base + 9);    q.wq = FP(base + 10);   q.wv = FP(base + 11);
  };
  auto fillV = [&](VPP& q, int base) {
    q.c1_b = FP(base + 0);  q.c1_w = FP(base + 1);  q.c2_b = FP(base + 2);
    q.c2_w = FP(base + 3);  q.gn1_b = FP(base + 4); q.gn1_s = FP(base + 5);
    q.gn2_b = FP(base + 6); q.gn2_s = FP(base + 7); q.lin_b = FP(base + 8);
    q.lin_w = FP(base + 9);
  };
  FFTP dec[4], enc[4]; VPP vp[3];
  for (int i = 0; i < 4; ++i) fillF(dec[i], PB + i * 12);
  const float* emb = FP(PB + 48);
  for (int i = 0; i < 4; ++i) fillF(enc[i], PB + 49 + i * 12);
  const float* mel_b = FP(PB + 97);
  const float* mel_w = FP(PB + 98);
  for (int j = 0; j < 3; ++j) fillV(vp[j], PB + 99 + j * 10);

  // workspace layout
  const size_t NE = (size_t)BB * MAXL * HD;             // 2,097,152 elements
  char* wsb = (char*)d_ws;
  auto balloc = [&](size_t bytes) { char* p = wsb; wsb += (bytes + 255) & ~(size_t)255; return p; };
  float*    X     = (float*)balloc(NE * 4);
  float*    Fb    = (float*)balloc(NE * 4);
  float*    Gb    = (float*)balloc(NE * 4);
  _Float16* X16   = (_Float16*)balloc(NE * 2);
  _Float16* QKV16 = (_Float16*)balloc(3 * NE * 2);      // fused (M, 768) QKV
  _Float16* O16   = (_Float16*)balloc(NE * 2);
  _Float16* H16   = (_Float16*)balloc((size_t)BB * MAXL * FFTF * 2);
  _Float16* Wp    = (_Float16*)balloc((size_t)9 * FFTF * HD * 2);  // max packed weight
  int*      cum   = (int*)balloc((size_t)BB * SS * 4);
  float*  gstats  = (float*)balloc(16 * 4);

  float* outf    = (float*)d_out;
  float* mel_out = outf;                                // (BB, MAXL, NMELS)
  float* dur_out = outf + (size_t)BB * MAXL * NMELS;    // (BB, SS)
  float* pit_out = dur_out + (size_t)BB * SS;
  float* ene_out = pit_out + (size_t)BB * SS;

  // 1) embedding
  embed_k<<<BB * SS, HD, 0, stream>>>(src_seq, emb, X, X16, HD);

  // 2) encoder
  for (int i = 0; i < 4; ++i)
    run_fft_block(stream, X, X16, SS, src_mask, enc[i], QKV16, O16, Fb, H16, Wp);

  // 3) variance predictors (duration, pitch, energy); QKV16 reused as f16 scratch
  run_vp(stream, X16, src_mask, vp[0], Fb, Gb, QKV16, gstats, dur_out, Wp);
  run_vp(stream, X16, src_mask, vp[1], Fb, Gb, QKV16, gstats, pit_out, Wp);
  run_vp(stream, X16, src_mask, vp[2], Fb, Gb, QKV16, gstats, ene_out, Wp);

  // 4) length regulate -> decoder input (f32 into Fb, f16 mirror directly into X16)
  cumsum_k<<<1, BB, 0, stream>>>(dur_tgt, cum, SS);
  length_reg_k<<<BB * MAXL, HD, 0, stream>>>(X, cum, Fb, X16, SS, HD, MAXL);
  hipMemcpyAsync(X, Fb, NE * sizeof(float), hipMemcpyDeviceToDevice, stream);

  // 5) decoder
  for (int i = 0; i < 4; ++i)
    run_fft_block(stream, X, X16, MAXL, mel_mask, dec[i], QKV16, O16, Fb, H16, Wp);

  // 6) mel projection (H -> 80) + bias
  run_gemm(stream, X16, mel_w, mel_b, mel_out, nullptr, Wp, BB * MAXL, NMELS, HD, MAXL,
           1, 0, 1L, (long)NMELS, 0L, 0);
}
